// ResidualBlock_41858751266868
// MI455X (gfx1250) — matrix operand order repaired and checked by reading, not compiled, not run on hardware
//
#include <hip/hip_runtime.h>

// ---------------------------------------------------------------------------
// CDNA5 (gfx1250) sparse residual block.
//   out = feat @ lin_w + conv2(relu(bn2(conv1(relu(bn1(feat)), W1))), W2)
// Matmuls on v_wmma_f32_16x16x32_bf16 (f32 accumulate).
// A-staging: GLOBAL_LOAD_ASYNC_TO_LDS_B128, software-pipelined (double-buffered
// LDS, in-order ASYNCcnt retirement => s_wait_asynccnt <= in-flight-batch).
// ---------------------------------------------------------------------------

#define GRID_G 100
#define CIN1   64
#define COUT1  128
#define BN_EPS 1e-4f

typedef __attribute__((ext_vector_type(16))) __bf16 v16bf;
typedef __attribute__((ext_vector_type(8)))  float  v8f;

struct alignas(16) U4 { unsigned int x, y, z, w; };

union FragBF16 { v16bf v; U4 u[2]; unsigned short s[16]; };
union FragAcc  { v8f  v; float f[8]; };

__device__ __forceinline__ unsigned short f32_to_bf16(float f) {
    unsigned int u = __float_as_uint(f);
    unsigned int r = (u + 0x7fffu + ((u >> 16) & 1u)) >> 16;   // round-to-nearest-even
    return (unsigned short)r;
}

// LDS byte offset of a __shared__ object (generic LDS pointers carry the
// LDS offset in the low 32 bits; aperture tag lives in the high bits)
__device__ __forceinline__ unsigned lds_off(const void* p) {
    return (unsigned)(unsigned long long)p;
}

// ---------------------------------------------------------------------------
// trivial fills (d_ws is poisoned; re-init every call for determinism)
// ---------------------------------------------------------------------------
__global__ void fill_i32_kernel(int* p, int val, int n) {
    int i = blockIdx.x * blockDim.x + threadIdx.x;
    if (i < n) p[i] = val;
}
__global__ void fill_f32_kernel(float* p, float val, int n) {
    int i = blockIdx.x * blockDim.x + threadIdx.x;
    if (i < n) p[i] = val;
}

// ---------------------------------------------------------------------------
// voxel map + neighbor table
// ---------------------------------------------------------------------------
__global__ void scatter_map_kernel(const int* __restrict__ pos, int* __restrict__ map, int N) {
    int n = blockIdx.x * blockDim.x + threadIdx.x;
    if (n >= N) return;
    int x = pos[n * 3 + 0], y = pos[n * 3 + 1], z = pos[n * 3 + 2];
    map[(x * GRID_G + y) * GRID_G + z] = n;
}

__global__ void build_nbr_kernel(const int* __restrict__ pos, const int* __restrict__ map,
                                 int* __restrict__ nbr, int N) {
    int i = blockIdx.x * blockDim.x + threadIdx.x;
    if (i >= N * 27) return;
    int n = i / 27, k = i % 27;
    int dx = k / 9 - 1, dy = (k / 3) % 3 - 1, dz = k % 3 - 1;
    int x = pos[n * 3 + 0] + dx;
    int y = pos[n * 3 + 1] + dy;
    int z = pos[n * 3 + 2] + dz;
    int v = -1;
    if (x >= 0 && x < GRID_G && y >= 0 && y < GRID_G && z >= 0 && z < GRID_G)
        v = map[(x * GRID_G + y) * GRID_G + z];
    nbr[i] = v;
}

// ---------------------------------------------------------------------------
// BN stats: per-channel sum / sumsq (block partials -> global atomics)
// ---------------------------------------------------------------------------
template <int C>
__global__ __launch_bounds__(256) void bn_stats_kernel(const float* __restrict__ x,
                                                       float* __restrict__ sums,
                                                       float* __restrict__ sumsq, int N) {
    constexpr int RPT = 256 / C;
    const int tid  = threadIdx.x;
    const int c    = tid % C;
    const int rsub = tid / C;
    const int r0   = blockIdx.x * 64;
    int rend = r0 + 64; if (rend > N) rend = N;

    float s = 0.f, s2 = 0.f;
    for (int r = r0 + rsub; r < rend; r += RPT) {
        float v = x[r * C + c];
        s += v; s2 += v * v;
    }
    __shared__ float buf[256];
    buf[tid] = s;
    __syncthreads();
    if (rsub == 0) {
        float t = buf[c];
        for (int i = 1; i < RPT; ++i) t += buf[i * C + c];
        atomicAdd(&sums[c], t);
    }
    __syncthreads();
    buf[tid] = s2;
    __syncthreads();
    if (rsub == 0) {
        float t = buf[c];
        for (int i = 1; i < RPT; ++i) t += buf[i * C + c];
        atomicAdd(&sumsq[c], t);
    }
}

__global__ void bn_finalize_kernel(const float* __restrict__ sums, const float* __restrict__ sumsq,
                                   float* __restrict__ mean, float* __restrict__ rstd, int C, int N) {
    int c = blockIdx.x * blockDim.x + threadIdx.x;
    if (c >= C) return;
    float m = sums[c] / (float)N;
    float v = sumsq[c] / (float)N - m * m;
    mean[c] = m;
    rstd[c] = rsqrtf(v + BN_EPS);
}

__global__ void bn_relu_bf16_kernel(const float* __restrict__ x, const float* __restrict__ mean,
                                    const float* __restrict__ rstd, const float* __restrict__ gamma,
                                    const float* __restrict__ beta, unsigned short* __restrict__ y,
                                    int total, int C) {
    int i = blockIdx.x * blockDim.x + threadIdx.x;
    if (i >= total) return;
    int c = i % C;
    float v = (x[i] - mean[c]) * rstd[c] * gamma[c] + beta[c];
    v = v > 0.f ? v : 0.f;
    y[i] = f32_to_bf16(v);
}

__global__ void to_bf16_kernel(const float* __restrict__ x, unsigned short* __restrict__ y, int total) {
    int i = blockIdx.x * blockDim.x + threadIdx.x;
    if (i < total) y[i] = f32_to_bf16(x[i]);
}

// ---------------------------------------------------------------------------
// Pre-swizzle weights into per-lane WMMA B-fragment order (bf16).
// dst linear index = (((k*T + t)*Q + q)*32 + lane)*16 + j
//   lane holds B column  col = t*16 + (lane&15)
//   element j holds      K   = q*32 + (lane>=16 ? 16 : 0) + j
// ---------------------------------------------------------------------------
__global__ void pack_weights_kernel(const float* __restrict__ W, unsigned short* __restrict__ dst,
                                    int taps, int cin, int cout) {
    int i = blockIdx.x * blockDim.x + threadIdx.x;
    int total = taps * cin * cout;
    if (i >= total) return;
    int Q = cin / 32, T = cout / 16;
    int j = i & 15;
    int l = (i >> 4) & 31;
    int rest = i >> 9;
    int q = rest % Q; rest /= Q;
    int t = rest % T;
    int k = rest / T;
    int kc  = q * 32 + ((l >= 16) ? 16 : 0) + j;
    int col = t * 16 + (l & 15);
    dst[i] = f32_to_bf16(W[(k * cin + kc) * cout + col]);
}

// ---------------------------------------------------------------------------
// WMMA conv kernel: block = 8 wave32s = one 32-point tile x all 128 Cout.
//   wave wt owns Cout columns [wt*16, wt*16+16); 2 row-halves share each B frag.
//   Software pipeline: while tap k is multiplied out of LDS buffer (k&1), the
//   async gather for tap k+1 streams into buffer ((k+1)&1). ASYNC loads retire
//   in order, so s_wait_asynccnt <= OPS (the in-flight next-tap batch) proves
//   tap k's rows have landed. Invalid neighbors read a zeroed row so EXEC
//   stays uniform (WMMA requires full EXEC) with no conditional zero-fill.
// ---------------------------------------------------------------------------
template <int CIN, int NTAPS, bool ADDBASE>
__global__ __launch_bounds__(256) void conv_wmma_kernel(const unsigned short* __restrict__ Abf,
                                                        const unsigned short* __restrict__ Wfrag,
                                                        const int* __restrict__ nbr,
                                                        const unsigned short* __restrict__ zrow,
                                                        float* __restrict__ out, int N) {
    constexpr int COUT = 128;
    constexpr int CT   = COUT / 16;   // 8 cout tiles == 8 waves
    constexpr int Q    = CIN / 32;    // K chunks per tap
    const int n0   = blockIdx.x * 32;
    const int tid  = threadIdx.x;
    const int lane = tid & 31;
    const int wt   = tid >> 5;

    __shared__ alignas(16) unsigned short Atile[2][32 * CIN];

    // issue the async gather of tap k into LDS buffer `buf` (no wait here)
    auto stage = [&](int k, unsigned short* buf) {
        const int m   = tid >> 3;   // row 0..31
        const int seg = tid & 7;
        const int n   = n0 + m;
        int src;
        if constexpr (NTAPS == 1) src = (n < N) ? n : -1;
        else                      src = (n < N) ? nbr[n * 27 + k] : -1;
        const unsigned short* g = (src >= 0) ? (Abf + (size_t)src * CIN) : zrow;
        if constexpr (CIN == 64) {
            const unsigned short* gp = g + seg * 8;                 // 16B / thread
            unsigned loff = lds_off(&buf[m * CIN + seg * 8]);
            asm volatile("global_load_async_to_lds_b128 %0, %1, off"
                         :: "v"(loff), "v"(gp) : "memory");
        } else {
            const unsigned short* gp = g + seg * 16;                // 32B / thread
            unsigned loff = lds_off(&buf[m * CIN + seg * 16]);
            asm volatile("global_load_async_to_lds_b128 %0, %1, off"
                         :: "v"(loff), "v"(gp) : "memory");
            const unsigned short* gp2 = gp + 8;
            unsigned loff2 = loff + 16;
            asm volatile("global_load_async_to_lds_b128 %0, %1, off"
                         :: "v"(loff2), "v"(gp2) : "memory");
        }
    };

    FragAcc acc0, acc1;
#pragma unroll
    for (int r = 0; r < 8; ++r) { acc0.f[r] = 0.f; acc1.f[r] = 0.f; }

    stage(0, Atile[0]);

    for (int k = 0; k < NTAPS; ++k) {
        const bool has_next = (k + 1 < NTAPS);
        if (has_next) stage(k + 1, Atile[(k + 1) & 1]);

        // prove tap k's batch retired (in-order): allow only the next batch
        // (1 async op/wave for CIN=64, 2 for CIN=128) to remain outstanding.
        if (has_next) {
            if constexpr (CIN == 64) asm volatile("s_wait_asynccnt 0x1" ::: "memory");
            else                     asm volatile("s_wait_asynccnt 0x2" ::: "memory");
        } else {
            asm volatile("s_wait_asynccnt 0x0" ::: "memory");
        }
        __syncthreads();   // all waves' tap-k rows visible

        const unsigned short* Ab = Atile[k & 1];
#pragma unroll
        for (int q = 0; q < Q; ++q) {
            // B fragment: one contiguous 32B chunk per lane (pre-swizzled)
            FragBF16 b;
            const unsigned short* bp = Wfrag + ((size_t)(((k * CT + wt) * Q + q) * 32 + lane) * 16);
            b.u[0] = *reinterpret_cast<const U4*>(bp);
            b.u[1] = *reinterpret_cast<const U4*>(bp + 8);
            if (has_next)   // warm L2/WGP$ for next tap's fragment
                __builtin_prefetch(bp + (size_t)CT * Q * 32 * 16, 0, 1);

            // A fragments (16x32, contiguous k order, same as packed B):
            //   lanes 0-15 : K = q*32 + [0..15]
            //   lanes 16-31: K = q*32 + [16..31]
            const int mrow = lane & 15;
            const int koff = q * 32 + ((lane >= 16) ? 16 : 0);
            FragBF16 a0, a1;
            a0.u[0] = *reinterpret_cast<const U4*>(&Ab[mrow * CIN + koff]);
            a0.u[1] = *reinterpret_cast<const U4*>(&Ab[mrow * CIN + koff + 8]);
            a1.u[0] = *reinterpret_cast<const U4*>(&Ab[(16 + mrow) * CIN + koff]);
            a1.u[1] = *reinterpret_cast<const U4*>(&Ab[(16 + mrow) * CIN + koff + 8]);

            acc0.v = __builtin_amdgcn_wmma_f32_16x16x32_bf16(
                false, a0.v, false, b.v, (short)0, acc0.v, false, false);
            acc1.v = __builtin_amdgcn_wmma_f32_16x16x32_bf16(
                false, a1.v, false, b.v, (short)0, acc1.v, false, false);
        }
        __syncthreads();   // readers done before this buffer is re-staged (k+2)
    }

    // C/D layout: VGPR r -> M = r + (lane>=16 ? 8 : 0), column = lane&15
    const int col   = wt * 16 + (lane & 15);
    const int mbase = (lane >= 16) ? 8 : 0;
#pragma unroll
    for (int r = 0; r < 8; ++r) {
        int n = n0 + mbase + r;
        if (n < N) {
            float v = acc0.f[r];
            if constexpr (ADDBASE) v += out[(size_t)n * COUT + col];
            out[(size_t)n * COUT + col] = v;
        }
        int n2 = n0 + 16 + mbase + r;
        if (n2 < N) {
            float v = acc1.f[r];
            if constexpr (ADDBASE) v += out[(size_t)n2 * COUT + col];
            out[(size_t)n2 * COUT + col] = v;
        }
    }
}

// ---------------------------------------------------------------------------
// host-side launch
// ---------------------------------------------------------------------------
static inline int ceil_div(int a, int b) { return (a + b - 1) / b; }

extern "C" void kernel_launch(void* const* d_in, const int* in_sizes, int n_in,
                              void* d_out, int out_size, void* d_ws, size_t ws_size,
                              hipStream_t stream) {
    const float* feat   = (const float*)d_in[0];
    const int*   pos    = (const int*)d_in[1];
    const float* lin_w  = (const float*)d_in[2];
    const float* gamma1 = (const float*)d_in[3];
    const float* beta1  = (const float*)d_in[4];
    const float* W1     = (const float*)d_in[5];
    const float* gamma2 = (const float*)d_in[6];
    const float* beta2  = (const float*)d_in[7];
    const float* W2     = (const float*)d_in[8];
    float* out = (float*)d_out;

    const int N = in_sizes[0] / CIN1;      // feat is [N, 64]
    const int MAPN = GRID_G * GRID_G * GRID_G;

    // ---- carve workspace (256B aligned chunks) ----
    char* p = (char*)d_ws;
    auto carve = [&](size_t bytes) -> void* {
        void* r = (void*)p;
        p += (bytes + 255) & ~(size_t)255;
        return r;
    };
    int*            map    = (int*)carve((size_t)MAPN * 4);
    int*            nbr    = (int*)carve((size_t)N * 27 * 4);
    float*          stats  = (float*)carve(768 * 4);
    unsigned short* zrow   = (unsigned short*)carve(256);            // zeroed dummy row
    unsigned short* featb  = (unsigned short*)carve((size_t)N * CIN1 * 2);
    unsigned short* h1b    = (unsigned short*)carve((size_t)N * CIN1 * 2);
    float*          h2     = (float*)carve((size_t)N * COUT1 * 4);
    unsigned short* h2b    = (unsigned short*)carve((size_t)N * COUT1 * 2);
    unsigned short* wB1    = (unsigned short*)carve((size_t)27 * CIN1 * COUT1 * 2);
    unsigned short* wB2    = (unsigned short*)carve((size_t)27 * COUT1 * COUT1 * 2);
    unsigned short* linB   = (unsigned short*)carve((size_t)CIN1 * COUT1 * 2);

    float* sums1 = stats, *sumsq1 = stats + 64,  *mean1 = stats + 128, *rstd1 = stats + 192;
    float* sums2 = stats + 256, *sumsq2 = stats + 384, *mean2 = stats + 512, *rstd2 = stats + 640;

    const int B = 256;

    // 1) init map + zero stats + zero dummy row (every call)
    fill_i32_kernel<<<ceil_div(MAPN, B), B, 0, stream>>>(map, -1, MAPN);
    fill_f32_kernel<<<ceil_div(768, B), B, 0, stream>>>(stats, 0.f, 768);
    fill_i32_kernel<<<1, 64, 0, stream>>>((int*)zrow, 0, 64);

    // 2) voxel map + neighbor table (reused by both convs)
    scatter_map_kernel<<<ceil_div(N, B), B, 0, stream>>>(pos, map, N);
    build_nbr_kernel<<<ceil_div(N * 27, B), B, 0, stream>>>(pos, map, nbr, N);

    // 3) pre-swizzle weights into WMMA B-fragment order (bf16)
    pack_weights_kernel<<<ceil_div(27 * CIN1 * COUT1, B), B, 0, stream>>>(W1, wB1, 27, CIN1, COUT1);
    pack_weights_kernel<<<ceil_div(27 * COUT1 * COUT1, B), B, 0, stream>>>(W2, wB2, 27, COUT1, COUT1);
    pack_weights_kernel<<<ceil_div(CIN1 * COUT1, B), B, 0, stream>>>(lin_w, linB, 1, CIN1, COUT1);

    // 4) feat -> bf16 (NiN shortcut operand)
    to_bf16_kernel<<<ceil_div(N * CIN1, B), B, 0, stream>>>(feat, featb, N * CIN1);

    // 5) BN1 + ReLU -> h1 (bf16)
    bn_stats_kernel<CIN1><<<ceil_div(N, 64), B, 0, stream>>>(feat, sums1, sumsq1, N);
    bn_finalize_kernel<<<1, 64, 0, stream>>>(sums1, sumsq1, mean1, rstd1, CIN1, N);
    bn_relu_bf16_kernel<<<ceil_div(N * CIN1, B), B, 0, stream>>>(feat, mean1, rstd1, gamma1, beta1,
                                                                 h1b, N * CIN1, CIN1);

    // 6) out = feat @ lin_w  (WMMA, identity gather)
    conv_wmma_kernel<CIN1, 1, false><<<ceil_div(N, 32), B, 0, stream>>>(featb, linB, nullptr, zrow, out, N);

    // 7) h2 = subm_conv(h1, W1)
    conv_wmma_kernel<CIN1, 27, false><<<ceil_div(N, 32), B, 0, stream>>>(h1b, wB1, nbr, zrow, h2, N);

    // 8) BN2 + ReLU -> h2 (bf16)
    bn_stats_kernel<COUT1><<<ceil_div(N, 64), B, 0, stream>>>(h2, sums2, sumsq2, N);
    bn_finalize_kernel<<<1, 128, 0, stream>>>(sums2, sumsq2, mean2, rstd2, COUT1, N);
    bn_relu_bf16_kernel<<<ceil_div(N * COUT1, B), B, 0, stream>>>(h2, mean2, rstd2, gamma2, beta2,
                                                                  h2b, N * COUT1, COUT1);

    // 9) out += subm_conv(h2, W2)
    conv_wmma_kernel<COUT1, 27, true><<<ceil_div(N, 32), B, 0, stream>>>(h2b, wB2, nbr, zrow, out, N);
}
